// MultiHeadAttention_v2_61976378081357
// MI455X (gfx1250) — matrix operand order repaired and checked by reading, not compiled, not run on hardware
//
#include <hip/hip_runtime.h>

#define Dd 128
#define Aa 128
#define Hh 4
#define Ff 128
#define Bb 64
#define Ll 200
#define L2k 50
#define DH 32
#define NEGF (-4294967296.0f)   // float(-2^32+1)

typedef __attribute__((ext_vector_type(16))) __bf16 v16bf;
typedef __attribute__((ext_vector_type(8)))  float  v8f;

__device__ __forceinline__ unsigned short f32_to_bf16(float f) {
  unsigned int u = __float_as_uint(f);
  unsigned int r = u + 0x7FFFu + ((u >> 16) & 1u);   // round-to-nearest-even
  return (unsigned short)(r >> 16);
}
__device__ __forceinline__ float bf16_to_f32(unsigned short h) {
  return __uint_as_float(((unsigned int)h) << 16);
}

// Async global->LDS copy of 16 bytes per lane (CDNA5 GLOBAL_LOAD_ASYNC_TO_LDS_B128).
// vdst = LDS byte address (generic addr low 32 bits == LDS offset per aperture rules).
__device__ __forceinline__ void async_copy_b128(const void* gsrc, void* ldst) {
  unsigned lds_addr = (unsigned)(size_t)ldst;
  unsigned long long gaddr = (unsigned long long)(size_t)gsrc;
  asm volatile("global_load_async_to_lds_b128 %0, %1, off"
               :: "v"(lds_addr), "v"(gaddr)
               : "memory");
}
__device__ __forceinline__ void async_wait0() {
  asm volatile("s_wait_asynccnt 0" ::: "memory");
}

union FragU { v16bf v; uint4 q[2]; };

// A-matrix 16x32 bf16 frag (ISA 7.12.2): lane half=0 -> K in {kb..kb+7, kb+16..kb+23}
__device__ __forceinline__ v16bf load_fragA(const unsigned short* base, int row0,
                                            int kb, int half, int lr) {
  const unsigned short* p = base + (row0 + lr) * 128 + kb + 8 * half;
  FragU f;
  f.q[0] = *reinterpret_cast<const uint4*>(p);        // K: kb+8h .. +7
  f.q[1] = *reinterpret_cast<const uint4*>(p + 16);   // K: kb+16+8h .. +7
  return f.v;
}
// B-matrix 32x16 bf16 frag: lane lr = column, K in {kb+8h..+7, kb+16+8h..+7} (matches A).
// Served from LDS-staged, pre-transposed weights Wt[N][K].
__device__ __forceinline__ v16bf load_fragB(const unsigned short* base, int n0,
                                            int kb, int half, int lr) {
  const unsigned short* p = base + (n0 + lr) * 128 + kb + 8 * half;
  FragU f;
  f.q[0] = *reinterpret_cast<const uint4*>(p);
  f.q[1] = *reinterpret_cast<const uint4*>(p + 16);
  return f.v;
}

__global__ void prep_weights_kernel(const float* __restrict__ Wk, const float* __restrict__ Wv,
                                    unsigned short* __restrict__ wk_t,
                                    unsigned short* __restrict__ wv_t) {
  int i = blockIdx.x * blockDim.x + threadIdx.x;     // i = n*128 + k
  if (i < 128 * 128) {
    int n = i >> 7, k = i & 127;
    wk_t[i] = f32_to_bf16(Wk[k * 128 + n]);          // transpose: Wt[n][k] = W[k][n]
    wv_t[i] = f32_to_bf16(Wv[k * 128 + n]);
  }
}

__global__ __launch_bounds__(256) void attn_kernel(
    const int* __restrict__ queries, const int* __restrict__ keys,
    const float* __restrict__ E,
    const float* __restrict__ Wq, const float* __restrict__ bq,
    const float* __restrict__ bk, const float* __restrict__ bv,
    const float* __restrict__ Wf, const float* __restrict__ bfv,
    const unsigned short* __restrict__ wkv_t,   // wk_t followed by wv_t (64 KB, contiguous)
    float* __restrict__ out) {
  __shared__ unsigned short kemb[64][128];     // 16 KB (rows 50..63 zero)
  __shared__ unsigned short Wlds[2][128][128]; // 64 KB staged bf16 weights [K|V][N][K]
  __shared__ float Kp[64][128];                // 32 KB
  __shared__ float Vp[64][128];                // 32 KB
  __shared__ float qe[128];
  __shared__ float Qv[128];
  __shared__ float ores[128];
  __shared__ float sc[4][64];
  __shared__ float rowsum[64];
  __shared__ int   kidx[64];

  const int t  = threadIdx.x;
  const int bl = blockIdx.x;                   // b*L + l

  // ---- Stage both weight matrices into LDS once per block, asynchronously
  // (overlaps with the data-dependent embedding gather below).
  {
    const uint4* wsrc = reinterpret_cast<const uint4*>(wkv_t);
    uint4* wdst = reinterpret_cast<uint4*>(&Wlds[0][0][0]);
#pragma unroll 4
    for (int i = t; i < 4096; i += 256)
      async_copy_b128(wsrc + i, wdst + i);
  }

  // ---- Phase A0: key indices
  if (t < 64) kidx[t] = (t < L2k) ? keys[(size_t)bl * L2k + t] : 0;
  __syncthreads();

  // ---- Phase A1: gather key embeddings (f32 -> bf16 LDS) and query embedding
  for (int i = t; i < L2k * 32; i += 256) {    // 50 rows x 32 float4
    int k = i >> 5, d4 = i & 31;
    const float4 e4 = reinterpret_cast<const float4*>(E + (size_t)kidx[k] * Dd)[d4];
    int d = d4 * 4;
    kemb[k][d + 0] = f32_to_bf16(e4.x);
    kemb[k][d + 1] = f32_to_bf16(e4.y);
    kemb[k][d + 2] = f32_to_bf16(e4.z);
    kemb[k][d + 3] = f32_to_bf16(e4.w);
  }
  for (int i = t; i < (64 - L2k) * 128; i += 256)
    (&kemb[L2k][0])[i] = 0;                    // zero-pad rows 50..63
  if (t < 32) {
    int qrow = queries[bl];
    const float4 e4 = reinterpret_cast<const float4*>(E + (size_t)qrow * Dd)[t];
    qe[t * 4 + 0] = e4.x; qe[t * 4 + 1] = e4.y;
    qe[t * 4 + 2] = e4.z; qe[t * 4 + 3] = e4.w;
  }
  async_wait0();                               // own async weight copies complete
  __syncthreads();                             // all waves' LDS writes visible

  // ---- Phase B: Q matvec (threads 0..127), key row-sums for mask (threads 128..191)
  if (t < 128) {
    float s = bq[t];
    for (int d = 0; d < Dd; ++d) s = fmaf(qe[d], Wq[d * Aa + t], s);
    Qv[t] = s;
  } else if (t < 192) {
    int k = t - 128;
    float s = 0.0f;
    if (k < L2k)
      for (int d = 0; d < Dd; ++d) s += bf16_to_f32(kemb[k][d]);
    rowsum[k] = s;
  }

  // ---- Phase C: K & V projections via bf16 WMMA (full EXEC, 8 waves)
  {
    const int wave = t >> 5;
    const int lane = t & 31;
    const int half = lane >> 4;
    const int lr   = lane & 15;
    const int sel  = wave >> 2;                // 0 = K-proj, 1 = V-proj
    const unsigned short* Wt = &Wlds[sel][0][0];
    const float* bias        = sel ? bv : bk;
    float* Pout              = sel ? &Vp[0][0] : &Kp[0][0];
    const int rt = (wave & 3) * 16;
#pragma unroll
    for (int ct = 0; ct < 8; ++ct) {
      const int n0 = ct * 16;
      const float bval = bias[n0 + lr];        // C/D layout: N = lane&15
      v8f acc;
#pragma unroll
      for (int j = 0; j < 8; ++j) acc[j] = bval;
#pragma unroll
      for (int ks = 0; ks < 4; ++ks) {
        v16bf am = load_fragA(&kemb[0][0], rt, ks * 32, half, lr);
        v16bf bm = load_fragB(Wt,          n0, ks * 32, half, lr);
        acc = __builtin_amdgcn_wmma_f32_16x16x32_bf16(
            false, am, false, bm, (short)0, acc, false, false);
      }
#pragma unroll
      for (int j = 0; j < 8; ++j)              // M = j + 8*half
        Pout[(rt + j + 8 * half) * 128 + n0 + lr] = acc[j];
    }
  }
  __syncthreads();

  // ---- Phase D: scores = Qh . Kh / sqrt(32), masked
  if (t < Hh * L2k) {
    int h = t / L2k, k = t % L2k;
    float s;
    if (rowsum[k] == 0.0f) {
      s = NEGF;
    } else {
      s = 0.0f;
      const int d0 = h * DH;
      for (int d = 0; d < DH; ++d) s = fmaf(Qv[d0 + d], Kp[k][d0 + d], s);
      s *= 0.17677669529663687f;               // 1/sqrt(32)
    }
    sc[h][k] = s;
  }
  __syncthreads();

  // ---- Phase E: per-head softmax over 50 keys
  if (t < Hh) {
    float m = -3.4e38f;
    for (int k = 0; k < L2k; ++k) m = fmaxf(m, sc[t][k]);
    float sum = 0.0f;
    for (int k = 0; k < L2k; ++k) { float e = __expf(sc[t][k] - m); sc[t][k] = e; sum += e; }
    float inv = 1.0f / sum;
    for (int k = 0; k < L2k; ++k) sc[t][k] *= inv;
  }
  __syncthreads();

  // ---- Phase F: attention-weighted V + residual
  if (t < Aa) {
    int h = t >> 5;
    float o = qe[t];
    for (int k = 0; k < L2k; ++k) o = fmaf(sc[h][k], Vp[k][t], o);
    ores[t] = o;
  }
  __syncthreads();

  // ---- Phase G: final projection
  if (t < Ff) {
    float r = bfv[t];
    for (int a = 0; a < Aa; ++a) r = fmaf(ores[a], Wf[a * Ff + t], r);
    out[(size_t)bl * Ff + t] = r;
  }
}

extern "C" void kernel_launch(void* const* d_in, const int* in_sizes, int n_in,
                              void* d_out, int out_size, void* d_ws, size_t ws_size,
                              hipStream_t stream) {
  const int*   queries = (const int*)d_in[0];
  const int*   keys    = (const int*)d_in[1];
  const float* E       = (const float*)d_in[2];
  const float* Wq      = (const float*)d_in[3];
  const float* bq      = (const float*)d_in[4];
  const float* Wk      = (const float*)d_in[5];
  const float* bk      = (const float*)d_in[6];
  const float* Wv      = (const float*)d_in[7];
  const float* bv      = (const float*)d_in[8];
  const float* Wf      = (const float*)d_in[9];
  const float* bfv     = (const float*)d_in[10];
  float* out = (float*)d_out;

  unsigned short* wk_t = (unsigned short*)d_ws;      // [128][128] bf16
  unsigned short* wv_t = wk_t + 128 * 128;           // contiguous behind wk_t

  prep_weights_kernel<<<64, 256, 0, stream>>>(Wk, Wv, wk_t, wv_t);
  attn_kernel<<<Bb * Ll, 256, 0, stream>>>(queries, keys, E, Wq, bq, bk, bv,
                                           Wf, bfv, wk_t, out);
}
